// Quantize_81449759801514
// MI455X (gfx1250) — matrix operand order repaired and checked by reading, not compiled, not run on hardware
//
#include <hip/hip_runtime.h>
#include <hip/hip_bf16.h>

typedef __attribute__((ext_vector_type(16))) _Float16 v16h;
typedef __attribute__((ext_vector_type(8)))  float    v8f;
typedef __attribute__((ext_vector_type(4)))  int      v4i;

#define DIMK   256
#define NCODE  8192
#define NROWS  32768
#define ROWS_PER_BLOCK 128
#define NWAVES 4
#define NTILES (NCODE / 16)
#define LDS_STRIDE (DIMK + 8)   // pad 8 halves (16B) -> lane stride 528B, distinct bank quads

// ---- CDNA5 async global->LDS path (gfx1250), with sync fallback ------------
#if defined(__gfx1250__) && __has_builtin(__builtin_amdgcn_global_load_async_to_lds_b128)
#define USE_ASYNC_LDS 1
#else
#define USE_ASYNC_LDS 0
#endif

#define GAS __attribute__((address_space(1)))
#define LAS __attribute__((address_space(3)))

__device__ __forceinline__ void wait_async_copies() {
#if USE_ASYNC_LDS
#if __has_builtin(__builtin_amdgcn_s_wait_asynccnt)
    __builtin_amdgcn_s_wait_asynccnt(0);
#else
    asm volatile("s_wait_asynccnt 0x0" ::: "memory");
#endif
#endif
}

// ---------------------------------------------------------------------------
// Prep: transpose codebook to code-major (K contiguous) in f16 + f32, and
// compute per-code squared norms. embed is [256][8192] f32.
// ---------------------------------------------------------------------------
__global__ void prep_embed_kernel(const float* __restrict__ embed,
                                  _Float16* __restrict__ eT16,
                                  float* __restrict__ eT32,
                                  float* __restrict__ e2) {
    const int j = blockIdx.x * blockDim.x + threadIdx.x;   // code id 0..8191
    float s = 0.0f;
    for (int d = 0; d < DIMK; ++d) {
        const float v = embed[(size_t)d * NCODE + j];      // coalesced across j
        eT32[(size_t)j * DIMK + d] = v;
        eT16[(size_t)j * DIMK + d] = (_Float16)v;
        s += v * v;
    }
    e2[j] = s;
}

// ---------------------------------------------------------------------------
// Main: WMMA distance GEMM + argmin + gather + commitment-loss partial.
// Block = 128 threads (4 waves). Wave owns 32 rows (two 16-row A frags kept
// resident in VGPRs for all 512 column tiles). 16-column B tiles double-
// buffered in LDS, filled via async global->LDS copies overlapped with WMMA.
// ---------------------------------------------------------------------------
__launch_bounds__(128, 1)
__global__ void vq_argmin_kernel(const float* __restrict__ x,      // [32768][256]
                                 const _Float16* __restrict__ eT16,// [8192][256]
                                 const float* __restrict__ eT32,   // [8192][256]
                                 const float* __restrict__ e2,     // [8192]
                                 float* __restrict__ outQ,         // [32768][256]
                                 float* __restrict__ outDiff,      // [1]
                                 float* __restrict__ outInd)       // [32768]
{
    __shared__ _Float16 ldsB[2][16 * LDS_STRIDE];   // double-buffered B tiles
    __shared__ int      ldsIdx[ROWS_PER_BLOCK];

    const int tid  = threadIdx.x;
    const int wave = tid >> 5;
    const int lane = tid & 31;
    const int l16  = lane & 15;
    const int half = lane >> 4;          // 0 or 1
    const int khiA = half * 16;          // A-frag: low half K0..15, high half K16..31
    const int kbB  = half * 16;          // B-frag: low half K0..15, high half K16..31

    const int m0      = blockIdx.x * ROWS_PER_BLOCK;
    const int rowBase = m0 + wave * 32;

    // ---- Load both A fragments once (f32 -> f16 inline) -------------------
    v16h a[2][8];
#pragma unroll
    for (int t = 0; t < 2; ++t) {
        const int row = rowBase + t * 16 + l16;          // A rows = lane%16
        const float* xr = x + (size_t)row * DIMK;
#pragma unroll
        for (int c = 0; c < 8; ++c) {
            const int k0 = c * 32;
#pragma unroll
            for (int h = 0; h < 8; ++h) {
                a[t][c][h]     = (_Float16)xr[k0 + khiA + h];
                a[t][c][8 + h] = (_Float16)xr[k0 + 8 + khiA + h];
            }
        }
    }

    float bestD[2][8];
    int   bestI[2][8];
#pragma unroll
    for (int t = 0; t < 2; ++t)
#pragma unroll
        for (int i = 0; i < 8; ++i) { bestD[t][i] = 3.4e38f; bestI[t][i] = 0; }

    // per-thread tile-copy assignment: 4 x 16B chunks of the 8KB tile
    auto issue_tile = [&](int tile, int buf) {
        const uint4* src = (const uint4*)(eT16 + (size_t)tile * 16 * DIMK);
#pragma unroll
        for (int i = 0; i < 4; ++i) {
            const int q   = tid + i * 128;   // 0..511 sixteen-byte chunks
            const int r   = q >> 5;          // code row in tile
            const int off = q & 31;          // 16B chunk within row
            _Float16* dst = &ldsB[buf][r * LDS_STRIDE + off * 8];
#if USE_ASYNC_LDS
            __builtin_amdgcn_global_load_async_to_lds_b128(
                (GAS v4i*)(src + q), (LAS v4i*)dst, 0, 0);
#else
            *(uint4*)dst = src[q];
#endif
        }
    };

    // ---- Sweep the codebook: double-buffered pipeline ---------------------
    issue_tile(0, 0);
    for (int t = 0; t < NTILES; ++t) {
        wait_async_copies();   // my chunks of tile t are in LDS (no-op in sync path)
        __syncthreads();       // everyone's tile-t chunks visible; buf[(t+1)&1] readers done
        if (t + 1 < NTILES) issue_tile(t + 1, (t + 1) & 1);

        v8f acc0 = {};
        v8f acc1 = {};
        const _Float16* brow = &ldsB[t & 1][l16 * LDS_STRIDE];  // B col = lane%16
#pragma unroll
        for (int c = 0; c < 8; ++c) {
            union { v16h v; uint4 q[2]; } bu;
            const uint4* bsrc = (const uint4*)(brow + c * 32 + kbB);
            bu.q[0] = bsrc[0];
            bu.q[1] = bsrc[1];
            acc0 = __builtin_amdgcn_wmma_f32_16x16x32_f16(
                false, a[0][c], false, bu.v, (short)0, acc0, false, false);
            acc1 = __builtin_amdgcn_wmma_f32_16x16x32_f16(
                false, a[1][c], false, bu.v, (short)0, acc1, false, false);
        }

        const int   col = t * 16 + l16;
        const float e2v = e2[col];
#pragma unroll
        for (int i = 0; i < 8; ++i) {
            const float d0 = e2v - 2.0f * acc0[i];   // ||x||^2 dropped: argmin-invariant
            if (d0 < bestD[0][i]) { bestD[0][i] = d0; bestI[0][i] = col; }
            const float d1 = e2v - 2.0f * acc1[i];
            if (d1 < bestD[1][i]) { bestD[1][i] = d1; bestI[1][i] = col; }
        }
    }

    // ---- Cross-lane argmin within each 16-lane half -----------------------
#pragma unroll
    for (int t = 0; t < 2; ++t) {
#pragma unroll
        for (int i = 0; i < 8; ++i) {
            float d  = bestD[t][i];
            int   ix = bestI[t][i];
#pragma unroll
            for (int m = 1; m < 16; m <<= 1) {
                const float od = __shfl_xor(d, m, 32);
                const int   oi = __shfl_xor(ix, m, 32);
                if (od < d || (od == d && oi < ix)) { d = od; ix = oi; }
            }
            if (l16 == 0) {
                // low half owns rows i, high half rows i+8 (C-matrix layout)
                ldsIdx[wave * 32 + t * 16 + i + half * 8] = ix;
            }
        }
    }
    __syncthreads();

    // ---- Gather quantize rows (f32 codebook) + commitment-loss partial ----
    float psum = 0.0f;
    for (int r = wave; r < ROWS_PER_BLOCK; r += NWAVES) {
        const int row = m0 + r;
        const int ix  = ldsIdx[r];
        const float4* qsrc = (const float4*)(eT32 + (size_t)ix  * DIMK);
        const float4* xsrc = (const float4*)(x    + (size_t)row * DIMK);
        float4*       qdst = (float4*)(outQ + (size_t)row * DIMK);
#pragma unroll
        for (int i = 0; i < 2; ++i) {
            const int e = lane + i * 32;     // 64 float4 per row
            const float4 q  = qsrc[e];
            const float4 xv = xsrc[e];
            qdst[e] = q;
            const float dx = q.x - xv.x, dy = q.y - xv.y;
            const float dz = q.z - xv.z, dw = q.w - xv.w;
            psum += dx * dx + dy * dy + dz * dz + dw * dw;
        }
        if (lane == 0) outInd[row] = (float)ix;
    }
#pragma unroll
    for (int m = 1; m < 32; m <<= 1) psum += __shfl_xor(psum, m, 32);
    if (lane == 0) atomicAdd(outDiff, psum * (1.0f / 8388608.0f));
}

// ---------------------------------------------------------------------------
extern "C" void kernel_launch(void* const* d_in, const int* in_sizes, int n_in,
                              void* d_out, int out_size, void* d_ws, size_t ws_size,
                              hipStream_t stream) {
    const float* input = (const float*)d_in[0];   // [8,4096,256] f32
    const float* embed = (const float*)d_in[1];   // [256,8192]   f32

    float* out     = (float*)d_out;
    float* outQ    = out;                          // 8388608 floats
    float* outDiff = out + (size_t)NROWS * DIMK;   // 1 float
    float* outInd  = outDiff + 1;                  // 32768 floats (indices)

    char* ws = (char*)d_ws;
    _Float16* eT16 = (_Float16*)ws;                                    // 4 MB
    float*    eT32 = (float*)(ws + (size_t)NCODE * DIMK * 2);          // 8 MB
    float*    e2   = (float*)(ws + (size_t)NCODE * DIMK * 2
                                 + (size_t)NCODE * DIMK * 4);          // 32 KB

    (void)hipMemsetAsync(outDiff, 0, sizeof(float), stream);   // capture-safe
    prep_embed_kernel<<<NCODE / 256, 256, 0, stream>>>(embed, eT16, eT32, e2);
    vq_argmin_kernel<<<NROWS / ROWS_PER_BLOCK, ROWS_PER_BLOCK, 0, stream>>>(
        input, eT16, eT32, e2, outQ, outDiff, outInd);
}
